// GATClassifier_11441792877139
// MI455X (gfx1250) — matrix operand order repaired and checked by reading, not compiled, not run on hardware
//
#include <hip/hip_runtime.h>
#include <cstdint>
#include <cstddef>

namespace {
constexpr int kN = 100000;   // nodes
constexpr int kE = 1000000;  // edges
constexpr int kH = 64;       // hidden
constexpr int kC = 10;       // classes
constexpr int kG = 64;       // graphs
}

typedef __attribute__((ext_vector_type(16))) _Float16 v16h;
typedef __attribute__((ext_vector_type(8)))  _Float16 v8h;
typedef __attribute__((ext_vector_type(8)))  float    v8f;

// order-preserving float<->int key for integer atomicMax on floats
__device__ __forceinline__ int fenc(float x) {
  int i = __float_as_int(x);
  return (i >= 0) ? i : (i ^ 0x7FFFFFFF);
}
__device__ __forceinline__ float fdec(int i) {
  return __int_as_float((i >= 0) ? i : (i ^ 0x7FFFFFFF));
}

// ---------------- utility fills ----------------
__global__ void fill_f32_kernel(float* __restrict__ p, float v, int n) {
  int i = blockIdx.x * blockDim.x + threadIdx.x;
  if (i < n) p[i] = v;
}
__global__ void fill_i32_kernel(int* __restrict__ p, int v, int n) {
  int i = blockIdx.x * blockDim.x + threadIdx.x;
  if (i < n) p[i] = v;
}

// ---------------- degree ----------------
__global__ void deg_kernel(const int* __restrict__ dst, float* __restrict__ deg, int e) {
  int i = blockIdx.x * blockDim.x + threadIdx.x;
  if (i < e) atomicAdd(&deg[dst[i]], 1.0f);
}

// ---------------- layer 1: feat = deg[:,None] * W1 row; el/er analytic ----------------
__global__ void layer1_feat_kernel(const float* __restrict__ deg, const float* __restrict__ W1,
                                   const float* __restrict__ al, const float* __restrict__ ar,
                                   float* __restrict__ feat, float* __restrict__ el,
                                   float* __restrict__ er, int n) {
  __shared__ float sw[kH], sal[kH], sar[kH];
  if (threadIdx.x < kH) {
    sw[threadIdx.x]  = W1[threadIdx.x];
    sal[threadIdx.x] = al[threadIdx.x];
    sar[threadIdx.x] = ar[threadIdx.x];
  }
  __syncthreads();
  int i = blockIdx.x * blockDim.x + threadIdx.x;
  if (i >= n) return;
  float d = deg[i];
  float dl = 0.f, dr = 0.f;
  float* f = feat + (size_t)i * kH;
  #pragma unroll
  for (int j = 0; j < kH; ++j) {
    float w = sw[j];
    f[j] = d * w;
    dl += w * sal[j];
    dr += w * sar[j];
  }
  el[i] = d * dl;
  er[i] = d * dr;
}

// ---------------- W (64x64 f32, row-major) -> transposed f16 copy ----------------
__global__ void convert_w_t_kernel(const float* __restrict__ W, _Float16* __restrict__ wt) {
  int idx = blockIdx.x * blockDim.x + threadIdx.x;
  if (idx >= kH * kH) return;
  int k = idx / kH, n = idx % kH;
  wt[n * kH + k] = (_Float16)W[idx];   // wt[n][k] = W[k][n]
}

// ---------------- WMMA GEMM: feat[n x 64] = h16[n x 64] @ W (wt is W^T in f16) ----------------
// One wave computes a full 16x64 row-block: A loaded once per K-chunk, 4 col tiles of W
// (8 KB, cache-resident) consumed by 4 accumulators -> 8 v_wmma per wave total.
__global__ __launch_bounds__(256)
void gemm_h_w_wmma_kernel(const _Float16* __restrict__ h16, const _Float16* __restrict__ wt,
                          float* __restrict__ feat, int nrows) {
  int gtid = blockIdx.x * blockDim.x + threadIdx.x;
  int wave = gtid >> 5;
  int lane = gtid & 31;
  int row0 = wave * 16;              // 16-row block of h
  if (row0 >= nrows) return;
  int half = lane >> 4;              // 0: lanes 0-15, 1: lanes 16-31
  int mn   = lane & 15;

  v8f acc[4] = {{}, {}, {}, {}};
  #pragma unroll
  for (int kc = 0; kc < kH; kc += 32) {
    // A tile 16x32 f16: lane(half,mn) needs K = kc + half*16 + [0..15]
    const _Float16* arow = h16 + (size_t)(row0 + mn) * kH + kc;
    v8h a_lo = *(const v8h*)(arow + half * 16);
    v8h a_hi = *(const v8h*)(arow + half * 16 + 8);
    v16h a = __builtin_shufflevector(a_lo, a_hi, 0,1,2,3,4,5,6,7,8,9,10,11,12,13,14,15);
    #pragma unroll
    for (int t = 0; t < 4; ++t) {
      // B tile 32x16 f16: lane needs column (t*16+mn), K = kc + half*16 + [0..15]; wt is W^T so contiguous
      const _Float16* bcol = wt + (size_t)(t * 16 + mn) * kH + kc + half * 16;
      v8h b_lo = *(const v8h*)(bcol);
      v8h b_hi = *(const v8h*)(bcol + 8);
      v16h b = __builtin_shufflevector(b_lo, b_hi, 0,1,2,3,4,5,6,7,8,9,10,11,12,13,14,15);
      acc[t] = __builtin_amdgcn_wmma_f32_16x16x32_f16(
          /*neg_a=*/false, a, /*neg_b=*/false, b,
          /*c_mod=*/(short)0, acc[t], /*reuse_a=*/false, /*reuse_b=*/false);
    }
  }
  // C/D layout: VGPR r -> M = half*8 + r, N = mn (within col tile t)
  #pragma unroll
  for (int t = 0; t < 4; ++t) {
    #pragma unroll
    for (int r = 0; r < 8; ++r) {
      feat[(size_t)(row0 + half * 8 + r) * kH + t * 16 + mn] = acc[t][r];
    }
  }
}

// ---------------- el = feat@al, er = feat@ar ----------------
__global__ void elr_kernel(const float* __restrict__ feat, const float* __restrict__ al,
                           const float* __restrict__ ar, float* __restrict__ el,
                           float* __restrict__ er, int n) {
  __shared__ float sal[kH], sar[kH];
  if (threadIdx.x < kH) { sal[threadIdx.x] = al[threadIdx.x]; sar[threadIdx.x] = ar[threadIdx.x]; }
  __syncthreads();
  int i = blockIdx.x * blockDim.x + threadIdx.x;
  if (i >= n) return;
  const float* f = feat + (size_t)i * kH;
  float a = 0.f, b = 0.f;
  #pragma unroll
  for (int j = 0; j < kH; ++j) { a += f[j] * sal[j]; b += f[j] * sar[j]; }
  el[i] = a;
  er[i] = b;
}

// ---------------- edge pass 1: e = leaky_relu, segment max via int-key atomicMax ----------------
__global__ void edge_max_kernel(const int* __restrict__ src, const int* __restrict__ dst,
                                const float* __restrict__ el, const float* __restrict__ er,
                                float* __restrict__ ebuf, int* __restrict__ m_i, int e) {
  int i = blockIdx.x * blockDim.x + threadIdx.x;
  if (i >= e) return;
  float x = el[src[i]] + er[dst[i]];
  x = (x > 0.f) ? x : 0.2f * x;
  ebuf[i] = x;
  atomicMax(&m_i[dst[i]], fenc(x));
}

// ---------------- edge pass 2: ex = exp(e - m[dst]); segment sum ----------------
__global__ void edge_expsum_kernel(const int* __restrict__ dst, const int* __restrict__ m_i,
                                   float* __restrict__ ebuf, float* __restrict__ s, int e) {
  int i = blockIdx.x * blockDim.x + threadIdx.x;
  if (i >= e) return;
  int d = dst[i];
  float ex = __expf(ebuf[i] - fdec(m_i[d]));
  ebuf[i] = ex;
  atomicAdd(&s[d], ex);
}

// ---------------- edge pass 3: out[dst] += (ex/s[dst]) * feat[src]  (wave per edge) ----------------
__global__ __launch_bounds__(256)
void edge_aggregate_kernel(const int* __restrict__ src, const int* __restrict__ dst,
                           const float* __restrict__ ebuf, const float* __restrict__ s,
                           const float* __restrict__ feat, float* __restrict__ outb, int e) {
  int t = blockIdx.x * blockDim.x + threadIdx.x;
  int i = t >> 5;          // edge
  int lane = t & 31;       // 2 features per lane, wave32 covers H=64
  if (i >= e) return;
  int si = src[i], di = dst[i];
  float alpha = ebuf[i] / s[di];
  const float2 v = ((const float2*)(feat + (size_t)si * kH))[lane];
  float* o = outb + (size_t)di * kH + 2 * lane;
  atomicAdd(o,     alpha * v.x);
  atomicAdd(o + 1, alpha * v.y);
}

// ---------------- h = relu(out + b); also emit f16 copy for next WMMA GEMM ----------------
__global__ void finish_layer_kernel(const float* __restrict__ acc, const float* __restrict__ bias,
                                    float* __restrict__ h, _Float16* __restrict__ h16, int total) {
  int idx = blockIdx.x * blockDim.x + threadIdx.x;
  if (idx >= total) return;
  float v = acc[idx] + bias[idx & (kH - 1)];
  v = (v > 0.f) ? v : 0.f;
  h[idx] = v;
  h16[idx] = (_Float16)v;
}

// ---------------- graph mean pool + classifier ----------------
__global__ void pool_count_kernel(const int* __restrict__ gid, float* __restrict__ counts, int n) {
  int i = blockIdx.x * blockDim.x + threadIdx.x;
  if (i < n) atomicAdd(&counts[gid[i]], 1.0f);
}
__global__ void pool_sum_kernel(const float* __restrict__ h, const int* __restrict__ gid,
                                float* __restrict__ hg, int n) {
  int idx = blockIdx.x * blockDim.x + threadIdx.x;
  if (idx >= n * kH) return;
  int i = idx >> 6, j = idx & 63;
  atomicAdd(&hg[gid[i] * kH + j], h[idx]);
}
__global__ void classify_kernel(const float* __restrict__ hg, const float* __restrict__ counts,
                                const float* __restrict__ Wc, const float* __restrict__ bc,
                                float* __restrict__ out) {
  int idx = blockIdx.x * blockDim.x + threadIdx.x;
  if (idx >= kG * kC) return;
  int g = idx / kC, c = idx % kC;
  float inv = 1.0f / fmaxf(counts[g], 1.0f);
  float acc = 0.f;
  #pragma unroll
  for (int k = 0; k < kH; ++k) acc += hg[g * kH + k] * Wc[k * kC + c];
  out[idx] = acc * inv + bc[c];
}

// =================================================================
extern "C" void kernel_launch(void* const* d_in, const int* in_sizes, int n_in,
                              void* d_out, int out_size, void* d_ws, size_t ws_size,
                              hipStream_t stream) {
  (void)in_sizes; (void)n_in; (void)out_size; (void)ws_size;
  const int*   src = (const int*)d_in[0];
  const int*   dst = (const int*)d_in[1];
  const int*   gid = (const int*)d_in[2];
  const float* W1  = (const float*)d_in[3];
  const float* al1 = (const float*)d_in[4];
  const float* ar1 = (const float*)d_in[5];
  const float* b1  = (const float*)d_in[6];
  const float* W2  = (const float*)d_in[7];
  const float* al2 = (const float*)d_in[8];
  const float* ar2 = (const float*)d_in[9];
  const float* b2  = (const float*)d_in[10];
  const float* W3  = (const float*)d_in[11];
  const float* al3 = (const float*)d_in[12];
  const float* ar3 = (const float*)d_in[13];
  const float* b3  = (const float*)d_in[14];
  const float* Wc  = (const float*)d_in[15];
  const float* bc  = (const float*)d_in[16];
  float* out = (float*)d_out;

  // ---- workspace layout (256B aligned regions) ----
  char* base = (char*)d_ws;
  auto carve = [&](size_t bytes) -> char* {
    char* p = base;
    base += (bytes + 255) & ~(size_t)255;
    return p;
  };
  float*    deg    = (float*)carve((size_t)kN * 4);
  float*    h      = (float*)carve((size_t)kN * kH * 4);
  _Float16* h16    = (_Float16*)carve((size_t)kN * kH * 2);
  float*    feat   = (float*)carve((size_t)kN * kH * 4);
  float*    outb   = (float*)carve((size_t)kN * kH * 4);
  float*    el     = (float*)carve((size_t)kN * 4);
  float*    er     = (float*)carve((size_t)kN * 4);
  int*      m_i    = (int*)carve((size_t)kN * 4);
  float*    sden   = (float*)carve((size_t)kN * 4);
  float*    ebuf   = (float*)carve((size_t)kE * 4);
  _Float16* w16t   = (_Float16*)carve((size_t)kH * kH * 2);
  float*    hg     = (float*)carve((size_t)kG * kH * 4);
  float*    counts = (float*)carve((size_t)kG * 4);

  constexpr int B = 256;
  auto blocks = [](long long n) { return (int)((n + B - 1) / B); };

  auto fillf = [&](float* p, float v, int n) {
    fill_f32_kernel<<<blocks(n), B, 0, stream>>>(p, v, n);
  };
  auto filli = [&](int* p, int v, int n) {
    fill_i32_kernel<<<blocks(n), B, 0, stream>>>(p, v, n);
  };

  // edge softmax + aggregation + bias/relu for one layer (feat/el/er already set)
  auto edge_phase = [&](const float* bias) {
    filli(m_i, INT_MIN, kN);
    edge_max_kernel<<<blocks(kE), B, 0, stream>>>(src, dst, el, er, ebuf, m_i, kE);
    fillf(sden, 0.f, kN);
    edge_expsum_kernel<<<blocks(kE), B, 0, stream>>>(dst, m_i, ebuf, sden, kE);
    fillf(outb, 0.f, kN * kH);
    edge_aggregate_kernel<<<blocks((long long)kE * 32), B, 0, stream>>>(src, dst, ebuf, sden, feat, outb, kE);
    finish_layer_kernel<<<blocks((long long)kN * kH), B, 0, stream>>>(outb, bias, h, h16, kN * kH);
  };

  // dense part for layers 2/3: feat = h @ W (WMMA), then el/er
  auto dense_phase = [&](const float* W, const float* al, const float* ar) {
    convert_w_t_kernel<<<blocks(kH * kH), B, 0, stream>>>(W, w16t);
    // one wave per 16-row block (computes all 4 col tiles)
    long long waves = (long long)(kN / 16);
    gemm_h_w_wmma_kernel<<<blocks(waves * 32), B, 0, stream>>>(h16, w16t, feat, kN);
    elr_kernel<<<blocks(kN), B, 0, stream>>>(feat, al, ar, el, er, kN);
  };

  // ---- degree ----
  fillf(deg, 0.f, kN);
  deg_kernel<<<blocks(kE), B, 0, stream>>>(dst, deg, kE);

  // ---- layer 1 (h is N x 1 = deg) ----
  layer1_feat_kernel<<<blocks(kN), B, 0, stream>>>(deg, W1, al1, ar1, feat, el, er, kN);
  edge_phase(b1);

  // ---- layer 2 ----
  dense_phase(W2, al2, ar2);
  edge_phase(b2);

  // ---- layer 3 ----
  dense_phase(W3, al3, ar3);
  edge_phase(b3);

  // ---- graph mean pool + classifier ----
  fillf(hg, 0.f, kG * kH);
  fillf(counts, 0.f, kG);
  pool_count_kernel<<<blocks(kN), B, 0, stream>>>(gid, counts, kN);
  pool_sum_kernel<<<blocks((long long)kN * kH), B, 0, stream>>>(h, gid, hg, kN);
  classify_kernel<<<1, kG * kC, 0, stream>>>(hg, counts, Wc, bc, out);
}
